// Proposed_Module_ClusterGNN_4638564679951
// MI455X (gfx1250) — matrix operand order repaired and checked by reading, not compiled, not run on hardware
//
#include <hip/hip_runtime.h>
#include <math.h>

// ---------------- WMMA types / wrapper (gfx1250, wave32) ----------------
typedef __attribute__((ext_vector_type(16))) _Float16 v16h;
typedef __attribute__((ext_vector_type(8)))  float    v8f;

__device__ __forceinline__ v8f wmma16(v16h a, v16h b, v8f c) {
  // D = A(16x32,f16) x B(32x16,f16) + C(16x16,f32)
  return __builtin_amdgcn_wmma_f32_16x16x32_f16(false, a, false, b, (short)0, c, false, false);
}

// A-matrix (16x32 f16) element->K map for lane/element (contiguous: k = 16*(lane/16) + e)
__device__ __forceinline__ int akmap(int lane, int e) {
  return ((lane >> 4) << 4) + e;
}

#define CLS 16      // clusters
#define NCl 256     // points per cluster
#define CH  32      // channels
#define C1O 64      // conv1 out
#define C2O 96      // conv2 out
#define ZF8 {0.f,0.f,0.f,0.f,0.f,0.f,0.f,0.f}

__device__ __forceinline__ float leaky(float x) { return x > 0.f ? x : 0.2f * x; }

// ---------------- zero stats accumulators ----------------
__global__ void k_zero(float* p) {
  int t = blockIdx.x * 256 + threadIdx.x;
  if (t < 6144) p[t] = 0.f;
}

// ---------------- pooling: (64^3,32) -> (4096,32) means over 4x4x4 ----------------
__global__ void k_pool(const float* __restrict__ xc, float* __restrict__ x) {
  int cell = blockIdx.x;           // 4096
  int c = threadIdx.x;             // 32
  int gx = cell >> 8, gy = (cell >> 4) & 15, gz = cell & 15;
  float s = 0.f;
  for (int fx = 0; fx < 4; ++fx)
    for (int fy = 0; fy < 4; ++fy)
      for (int fz = 0; fz < 4; ++fz)
        s += xc[((((gx*4+fx)*64) + (gy*4+fy))*64 + (gz*4+fz))*CH + c];
  x[cell*CH + c] = s * (1.f/64.f);
}

// ---------------- stable sort by counting + gather sorted xs ----------------
__global__ void k_sort(const int* __restrict__ ci, const float* __restrict__ x,
                       int* __restrict__ sidx, float* __restrict__ xs) {
  int j = blockIdx.x * 256 + threadIdx.x;
  if (j >= 4096) return;
  int k = ci[j];
  int pos = 0;
  for (int q = 0; q < j; ++q) pos += (ci[q] == k) ? 1 : 0;
  int d = k * NCl + pos;
  sidx[d] = j;
  for (int c = 0; c < CH; ++c) xs[d*CH + c] = x[j*CH + c];
}

// ---------------- per-cluster centroid MLP -> sw, and Wg = gcn_w @ sw ----------------
__global__ __launch_bounds__(256)
void k_prep(const float* __restrict__ xs,
            const float* __restrict__ w1, const float* __restrict__ b1,
            const float* __restrict__ w2, const float* __restrict__ b2,
            const float* __restrict__ w3, const float* __restrict__ b3,
            const float* __restrict__ gcn, float* __restrict__ wgout) {
  __shared__ float cent[32], h1[128], h2[256], sw[1024];
  int cl = blockIdx.x, t = threadIdx.x;
  if (t < 32) {
    float s = 0.f;
    for (int r = 0; r < NCl; ++r) s += xs[cl*NCl*CH + r*CH + t];
    cent[t] = s * (1.f/256.f);
  }
  __syncthreads();
  if (t < 128) {
    float s = b1[t];
    for (int c = 0; c < 32; ++c) s += cent[c] * w1[c*128 + t];
    h1[t] = fmaxf(s, 0.f);
  }
  __syncthreads();
  {
    float s = b2[t];
    for (int k = 0; k < 128; ++k) s += h1[k] * w2[k*256 + t];
    h2[t] = fmaxf(s, 0.f);
  }
  __syncthreads();
  for (int q = 0; q < 4; ++q) {
    int o = t*4 + q;
    float s = b3[o];
    for (int k = 0; k < 256; ++k) s += h2[k] * w3[k*1024 + o];
    sw[o] = 1.f / (1.f + expf(-s));
  }
  __syncthreads();
  for (int q = 0; q < 8; ++q) {
    int idx = t*8 + q;          // 0..2047 : (a=in-ch 0..63, c=out-ch 0..31)
    int a = idx >> 5, c = idx & 31;
    float s = 0.f;
    for (int k = 0; k < 32; ++k) s += gcn[cl*2048 + a*32 + k] * sw[k*32 + c];
    wgout[cl*2048 + idx] = s;
  }
}

// ---------------- pass 1: conv1 stats (sum/sumsq per channel) ----------------
__global__ __launch_bounds__(256)
void k_stats1(const float* __restrict__ xs, const float* __restrict__ w1g,
              const float* __restrict__ b1g, float* __restrict__ ssumg, float* __restrict__ ssqg) {
  __shared__ _Float16 lxs[NCl*CH];      // 16KB
  int cl = blockIdx.x, tid = threadIdx.x;
  for (int i = tid; i < NCl*CH; i += 256) lxs[i] = (_Float16)xs[cl*NCl*CH + i];
  __syncthreads();
  int lane = tid & 31, w = tid >> 5;
  int n = lane & 15, koff = (lane >> 4) << 4;
  const float* W1 = w1g + cl*C1O*CH;
  v16h B1[4];
#pragma unroll
  for (int nt = 0; nt < 4; ++nt)
#pragma unroll
    for (int e = 0; e < 16; ++e) B1[nt][e] = (_Float16)W1[(nt*16+n)*CH + koff + e];
  float b1v[4];
#pragma unroll
  for (int nt = 0; nt < 4; ++nt) b1v[nt] = b1g[cl*C1O + nt*16 + n];
  float ssum[4] = {0,0,0,0}, ssq[4] = {0,0,0,0};
  int gw = blockIdx.y*8 + w;            // 0..63
  for (int t = gw*64; t < gw*64 + 64; ++t) {   // 4096 tiles per cluster total
    int i = t >> 4, jt = t & 15;
    const _Float16* xr = &lxs[i*CH];
    const _Float16* xc = &lxs[(jt*16 + (lane & 15))*CH];
    v16h A;
#pragma unroll
    for (int e = 0; e < 16; ++e) {
      int c = akmap(lane, e);
      A[e] = (_Float16)fabsf((float)xr[c] - (float)xc[c]);
    }
#pragma unroll
    for (int nt = 0; nt < 4; ++nt) {
      v8f acc = ZF8;
      acc = wmma16(A, B1[nt], acc);
#pragma unroll
      for (int r = 0; r < 8; ++r) { float v = acc[r] + b1v[nt]; ssum[nt] += v; ssq[nt] += v*v; }
    }
  }
#pragma unroll
  for (int nt = 0; nt < 4; ++nt) {
    int o = nt*16 + n;
    atomicAdd(&ssumg[cl*C1O + o], ssum[nt]);
    atomicAdd(&ssqg [cl*C1O + o], ssq [nt]);
  }
}

// ---------------- derive BN scale/bias from sums ----------------
__global__ void k_bnparams(const float* __restrict__ sum, const float* __restrict__ sq,
                           const float* __restrict__ g, const float* __restrict__ b,
                           float* __restrict__ sb, int per, float invN) {
  int t = blockIdx.x * 256 + threadIdx.x;
  if (t < CLS * per) {
    float m = sum[t] * invN;
    float v = sq[t] * invN - m*m;
    float s = g[t] * rsqrtf(v + 1e-5f);
    sb[t] = s;
    sb[CLS*per + t] = b[t] - m*s;
  }
}

// ---------------- pass 2: conv1->bn1->leaky->conv2 stats ----------------
__global__ __launch_bounds__(256)
void k_stats2(const float* __restrict__ xs, const float* __restrict__ w1g,
              const float* __restrict__ b1g, const float* __restrict__ sb1,
              const float* __restrict__ w2g, const float* __restrict__ b2g,
              float* __restrict__ ssumg, float* __restrict__ ssqg) {
  __shared__ _Float16 lxs[NCl*CH];      // 16KB
  __shared__ _Float16 la1[8*16*C1O];    // 16KB per-wave a1 scratch
  int cl = blockIdx.x, tid = threadIdx.x;
  for (int i = tid; i < NCl*CH; i += 256) lxs[i] = (_Float16)xs[cl*NCl*CH + i];
  __syncthreads();
  int lane = tid & 31, w = tid >> 5;
  int n = lane & 15, koff = (lane >> 4) << 4;
  const float* W1 = w1g + cl*C1O*CH;
  const float* W2 = w2g + cl*C2O*C1O;
  v16h B1[4];
#pragma unroll
  for (int nt = 0; nt < 4; ++nt)
#pragma unroll
    for (int e = 0; e < 16; ++e) B1[nt][e] = (_Float16)W1[(nt*16+n)*CH + koff + e];
  v16h B2[6][2];
#pragma unroll
  for (int nt2 = 0; nt2 < 6; ++nt2)
#pragma unroll
    for (int kc = 0; kc < 2; ++kc)
#pragma unroll
      for (int e = 0; e < 16; ++e)
        B2[nt2][kc][e] = (_Float16)W2[(nt2*16+n)*C1O + kc*32 + koff + e];
  float b1v[4], s1v[4], bb1v[4];
#pragma unroll
  for (int nt = 0; nt < 4; ++nt) {
    int o = nt*16 + n;
    b1v[nt] = b1g[cl*C1O + o];
    s1v[nt] = sb1[cl*C1O + o];
    bb1v[nt] = sb1[CLS*C1O + cl*C1O + o];
  }
  float b2v[6];
#pragma unroll
  for (int nt2 = 0; nt2 < 6; ++nt2) b2v[nt2] = b2g[cl*C2O + nt2*16 + n];
  float ssum[6] = {0,0,0,0,0,0}, ssq[6] = {0,0,0,0,0,0};
  _Float16* myA1 = &la1[w * 16 * C1O];
  int gw = blockIdx.y*8 + w;
  for (int t = gw*64; t < gw*64 + 64; ++t) {
    int i = t >> 4, jt = t & 15;
    const _Float16* xr = &lxs[i*CH];
    const _Float16* xc = &lxs[(jt*16 + (lane & 15))*CH];
    v16h A;
#pragma unroll
    for (int e = 0; e < 16; ++e) {
      int c = akmap(lane, e);
      A[e] = (_Float16)fabsf((float)xr[c] - (float)xc[c]);
    }
#pragma unroll
    for (int nt = 0; nt < 4; ++nt) {
      v8f acc = ZF8;
      acc = wmma16(A, B1[nt], acc);
#pragma unroll
      for (int r = 0; r < 8; ++r) {
        float v = (acc[r] + b1v[nt]) * s1v[nt] + bb1v[nt];
        v = leaky(v);
        int M = r + ((lane >> 4) << 3);
        myA1[M*C1O + nt*16 + n] = (_Float16)v;
      }
    }
    asm volatile("s_wait_dscnt 0" ::: "memory");   // order in-wave LDS store->load
    v16h A2[2];
#pragma unroll
    for (int kc = 0; kc < 2; ++kc)
#pragma unroll
      for (int e = 0; e < 16; ++e) {
        int c = kc*32 + akmap(lane, e);
        A2[kc][e] = myA1[(lane & 15)*C1O + c];
      }
#pragma unroll
    for (int nt2 = 0; nt2 < 6; ++nt2) {
      v8f acc = ZF8;
      acc = wmma16(A2[0], B2[nt2][0], acc);
      acc = wmma16(A2[1], B2[nt2][1], acc);
#pragma unroll
      for (int r = 0; r < 8; ++r) { float v = acc[r] + b2v[nt2]; ssum[nt2] += v; ssq[nt2] += v*v; }
    }
  }
#pragma unroll
  for (int nt2 = 0; nt2 < 6; ++nt2) {
    int o = nt2*16 + n;
    atomicAdd(&ssumg[cl*C2O + o], ssum[nt2]);
    atomicAdd(&ssqg [cl*C2O + o], ssq [nt2]);
  }
}

// ---------------- main pass: full pipeline per (cluster, 16-row block) ----------------
__global__ __launch_bounds__(256)
void k_main(const float* __restrict__ xs, const float* __restrict__ w1g,
            const float* __restrict__ b1g, const float* __restrict__ sb1,
            const float* __restrict__ w2g, const float* __restrict__ b2g,
            const float* __restrict__ sb2, const float* __restrict__ w3g,
            const float* __restrict__ b3g, const float* __restrict__ wgg,
            float* __restrict__ hout, float* __restrict__ s3sum, float* __restrict__ s3sq) {
  __shared__ _Float16 lxs[NCl*CH];    // 16KB
  __shared__ float    llog[16*256];   // 16KB logits -> adj
  __shared__ float    lscr[4096];     // 16KB: per-wave a1 (f16 view) / reductions / adjx
  int cl = blockIdx.x, ib = blockIdx.y;   // ib: 16-row block 0..15
  int tid = threadIdx.x;
  for (int i = tid; i < NCl*CH; i += 256) lxs[i] = (_Float16)xs[cl*NCl*CH + i];
  __syncthreads();
  int lane = tid & 31, w = tid >> 5;
  int n = lane & 15, koff = (lane >> 4) << 4;
  const float* W1 = w1g + cl*C1O*CH;
  const float* W2 = w2g + cl*C2O*C1O;
  v16h B1[4];
#pragma unroll
  for (int nt = 0; nt < 4; ++nt)
#pragma unroll
    for (int e = 0; e < 16; ++e) B1[nt][e] = (_Float16)W1[(nt*16+n)*CH + koff + e];
  v16h B2[6][2];
#pragma unroll
  for (int nt2 = 0; nt2 < 6; ++nt2)
#pragma unroll
    for (int kc = 0; kc < 2; ++kc)
#pragma unroll
      for (int e = 0; e < 16; ++e)
        B2[nt2][kc][e] = (_Float16)W2[(nt2*16+n)*C1O + kc*32 + koff + e];
  float b1v[4], s1v[4], bb1v[4];
#pragma unroll
  for (int nt = 0; nt < 4; ++nt) {
    int o = nt*16 + n;
    b1v[nt] = b1g[cl*C1O + o]; s1v[nt] = sb1[cl*C1O + o]; bb1v[nt] = sb1[CLS*C1O + cl*C1O + o];
  }
  float b2v[6], s2v[6], bb2v[6], w3v[6];
#pragma unroll
  for (int nt2 = 0; nt2 < 6; ++nt2) {
    int o = nt2*16 + n;
    b2v[nt2] = b2g[cl*C2O + o]; s2v[nt2] = sb2[cl*C2O + o];
    bb2v[nt2] = sb2[CLS*C2O + cl*C2O + o]; w3v[nt2] = w3g[cl*C2O + o];
  }
  float b3s = b3g[cl];
  _Float16* myA1 = (_Float16*)(lscr + w*512);    // 16x64 f16 per wave

  for (int tt = 0; tt < 32; ++tt) {              // uniform trip count across waves
    int t = tt*8 + w;                            // 0..255 : (il 0..15, jt 0..15)
    int il = t >> 4, jt = t & 15;
    int i = ib*16 + il;                          // row within cluster
    const _Float16* xr = &lxs[i*CH];
    const _Float16* xc = &lxs[(jt*16 + (lane & 15))*CH];
    v16h A;
#pragma unroll
    for (int e = 0; e < 16; ++e) {
      int c = akmap(lane, e);
      A[e] = (_Float16)fabsf((float)xr[c] - (float)xc[c]);
    }
#pragma unroll
    for (int nt = 0; nt < 4; ++nt) {
      v8f acc = ZF8;
      acc = wmma16(A, B1[nt], acc);
#pragma unroll
      for (int r = 0; r < 8; ++r) {
        float v = (acc[r] + b1v[nt]) * s1v[nt] + bb1v[nt];
        v = leaky(v);
        int M = r + ((lane >> 4) << 3);
        myA1[M*C1O + nt*16 + n] = (_Float16)v;
      }
    }
    asm volatile("s_wait_dscnt 0" ::: "memory");
    v16h A2[2];
#pragma unroll
    for (int kc = 0; kc < 2; ++kc)
#pragma unroll
      for (int e = 0; e < 16; ++e) {
        int c = kc*32 + akmap(lane, e);
        A2[kc][e] = myA1[(lane & 15)*C1O + c];
      }
    float part[8] = {0,0,0,0,0,0,0,0};
#pragma unroll
    for (int nt2 = 0; nt2 < 6; ++nt2) {
      v8f acc = ZF8;
      acc = wmma16(A2[0], B2[nt2][0], acc);
      acc = wmma16(A2[1], B2[nt2][1], acc);
#pragma unroll
      for (int r = 0; r < 8; ++r) {
        float v = (acc[r] + b2v[nt2]) * s2v[nt2] + bb2v[nt2];
        v = leaky(v);
        part[r] += v * w3v[nt2];
      }
    }
    // reduce logits across the 16 lanes of each half-wave
#pragma unroll
    for (int mask = 1; mask < 16; mask <<= 1)
#pragma unroll
      for (int r = 0; r < 8; ++r) part[r] += __shfl_xor(part[r], mask, 32);
    if ((lane & 15) == 0) {
      int mb = (lane >> 4) << 3;
#pragma unroll
      for (int r = 0; r < 8; ++r)
        llog[il*256 + jt*16 + mb + r] = part[r] + b3s;
    }
  }
  __syncthreads();

  // ---- softmax over 256 columns for 16 rows (thread = (row, 16-col segment)) ----
  int row = tid >> 4, seg = tid & 15;
  int ci = ib*16 + row;                          // diagonal column (within cluster)
  float* lr = &llog[row*256];
  float pm = -1e30f;
  for (int q = 0; q < 16; ++q) {
    int c = seg*16 + q;
    float v = lr[c] - (c == ci ? 1e8f : 0.f);
    pm = fmaxf(pm, v);
  }
  lscr[tid] = pm;
  __syncthreads();
  float rm = -1e30f;
  for (int q = 0; q < 16; ++q) rm = fmaxf(rm, lscr[row*16 + q]);
  __syncthreads();
  float ps = 0.f;
  for (int q = 0; q < 16; ++q) {
    int c = seg*16 + q;
    float v = lr[c] - (c == ci ? 1e8f : 0.f);
    float e = expf(v - rm);
    lr[c] = e;
    ps += e;
  }
  lscr[tid] = ps;
  __syncthreads();
  float rs = 0.f;
  for (int q = 0; q < 16; ++q) rs += lscr[row*16 + q];
  float inv = 1.f / rs;
  for (int q = 0; q < 16; ++q) lr[seg*16 + q] *= inv;
  __syncthreads();

  // ---- adjx = adj(16x256) @ xi(256x32), waves 0/1 handle 16-ch tiles ----
  if (w < 2) {
    int nt = w;
    v8f acc = ZF8;
    for (int kc = 0; kc < 8; ++kc) {
      v16h Aa, Bb;
#pragma unroll
      for (int e = 0; e < 16; ++e) {
        int kk = akmap(lane, e);
        Aa[e] = (_Float16)llog[(lane & 15)*256 + kc*32 + kk];
      }
#pragma unroll
      for (int e = 0; e < 16; ++e) {
        int jj = kc*32 + koff + e;
        Bb[e] = lxs[jj*CH + nt*16 + n];
      }
      acc = wmma16(Aa, Bb, acc);
    }
#pragma unroll
    for (int r = 0; r < 8; ++r) {
      int M = r + ((lane >> 4) << 3);
      lscr[M*32 + nt*16 + n] = acc[r];
    }
  }
  __syncthreads();

  // ---- h = leaky( [xi, adjx](16x64) @ Wg(64x32) ), wave 0 ----
  if (w == 0) {
    v16h A0, A1;
#pragma unroll
    for (int e = 0; e < 16; ++e) {
      int c = akmap(lane, e);
      A0[e] = lxs[(ib*16 + (lane & 15))*CH + c];
      A1[e] = (_Float16)lscr[(lane & 15)*32 + c];
    }
    const float* Wg = &wgg[cl*2048];
#pragma unroll
    for (int nt = 0; nt < 2; ++nt) {
      v16h B0, Bh;
#pragma unroll
      for (int e = 0; e < 16; ++e) {
        int k0 = koff + e;
        B0[e] = (_Float16)Wg[k0*32 + nt*16 + n];
        Bh[e] = (_Float16)Wg[(32 + k0)*32 + nt*16 + n];
      }
      v8f acc = ZF8;
      acc = wmma16(A0, B0, acc);
      acc = wmma16(A1, Bh, acc);
#pragma unroll
      for (int r = 0; r < 8; ++r) {
        float v = leaky(acc[r]);
        int M = r + ((lane >> 4) << 3);
        int g = ib*16 + M;
        hout[cl*NCl*CH + g*CH + nt*16 + n] = v;
        atomicAdd(&s3sum[cl*CH + nt*16 + n], v);
        atomicAdd(&s3sq [cl*CH + nt*16 + n], v*v);
      }
    }
  }
}

// ---------------- bn3 + scatter back through sorted index ----------------
__global__ void k_scatter(const float* __restrict__ h, const float* __restrict__ sb3,
                          const int* __restrict__ sidx, float* __restrict__ xg) {
  int cl = blockIdx.x, r = threadIdx.x;
  int orig = sidx[cl*NCl + r];
  for (int c = 0; c < CH; ++c) {
    float v = h[cl*NCl*CH + r*CH + c];
    v = v * sb3[cl*CH + c] + sb3[CLS*CH + cl*CH + c];
    xg[orig*CH + c] = v;
  }
}

// ---------------- unfold (pad 1,2 / 4-tap) + residual ----------------
__global__ __launch_bounds__(256)
void k_out(const float* __restrict__ xc, const float* __restrict__ xg, float* __restrict__ out) {
  int t = blockIdx.x * 256 + threadIdx.x;   // 2,097,152 threads, float4 each
  int c4 = (t & 7) << 2;
  int v = t >> 3;
  int Z = v & 63; v >>= 6;
  int Y = v & 63; int X = v >> 6;
  int cx = (X >> 2) + (X & 3) - 1;
  int cy = (Y >> 2) + (Y & 3) - 1;
  int cz = (Z >> 2) + (Z & 3) - 1;
  int idx = ((X*64 + Y)*64 + Z)*CH + c4;
  float4 a = *(const float4*)(xc + idx);
  if ((unsigned)cx < 16u && (unsigned)cy < 16u && (unsigned)cz < 16u) {
    float4 g = *(const float4*)(xg + ((cx*16 + cy)*16 + cz)*CH + c4);
    a.x += g.x; a.y += g.y; a.z += g.z; a.w += g.w;
  }
  *(float4*)(out + idx) = a;
}

// ---------------- host-side launch ----------------
extern "C" void kernel_launch(void* const* d_in, const int* in_sizes, int n_in,
                              void* d_out, int out_size, void* d_ws, size_t ws_size,
                              hipStream_t stream) {
  const float* x_concat = (const float*)d_in[0];
  const int*   ci       = (const int*)d_in[1];
  const float* conv1_w  = (const float*)d_in[2];
  const float* conv1_b  = (const float*)d_in[3];
  const float* bn1_g    = (const float*)d_in[4];
  const float* bn1_b    = (const float*)d_in[5];
  const float* conv2_w  = (const float*)d_in[6];
  const float* conv2_b  = (const float*)d_in[7];
  const float* bn2_g    = (const float*)d_in[8];
  const float* bn2_b    = (const float*)d_in[9];
  const float* conv3_w  = (const float*)d_in[10];
  const float* conv3_b  = (const float*)d_in[11];
  const float* gcn_w    = (const float*)d_in[12];
  const float* bn3_g    = (const float*)d_in[13];
  const float* bn3_b    = (const float*)d_in[14];
  const float* mlp_w1   = (const float*)d_in[15];
  const float* mlp_b1   = (const float*)d_in[16];
  const float* mlp_w2   = (const float*)d_in[17];
  const float* mlp_b2   = (const float*)d_in[18];
  const float* mlp_w3   = (const float*)d_in[19];
  const float* mlp_b3   = (const float*)d_in[20];

  float* ws   = (float*)d_ws;
  float* xbuf = ws;                        // 131072 f
  float* xs   = xbuf + 131072;             // 131072 f
  int*   sidx = (int*)(xs + 131072);       // 4096 i
  float* wg   = (float*)(sidx + 4096);     // 32768 f
  float* st   = wg + 32768;                // 6144 f (stat accumulators)
  float* sum1 = st;          float* sq1 = st + 1024;
  float* sum2 = st + 2048;   float* sq2 = st + 3584;
  float* sum3 = st + 5120;   float* sq3 = st + 5632;
  float* sb1  = st + 6144;                 // 2048 f (scale|bias)
  float* sb2  = sb1 + 2048;                // 3072 f
  float* sb3  = sb2 + 3072;                // 1024 f
  float* hbuf = sb3 + 1024;                // 131072 f
  float* xg   = hbuf + 131072;             // 131072 f

  k_zero<<<24, 256, 0, stream>>>(st);
  k_pool<<<4096, 32, 0, stream>>>(x_concat, xbuf);
  k_sort<<<16, 256, 0, stream>>>(ci, xbuf, sidx, xs);
  k_prep<<<16, 256, 0, stream>>>(xs, mlp_w1, mlp_b1, mlp_w2, mlp_b2, mlp_w3, mlp_b3, gcn_w, wg);
  k_stats1<<<dim3(16, 8), 256, 0, stream>>>(xs, conv1_w, conv1_b, sum1, sq1);
  k_bnparams<<<4, 256, 0, stream>>>(sum1, sq1, bn1_g, bn1_b, sb1, 64, 1.f/65536.f);
  k_stats2<<<dim3(16, 8), 256, 0, stream>>>(xs, conv1_w, conv1_b, sb1, conv2_w, conv2_b, sum2, sq2);
  k_bnparams<<<6, 256, 0, stream>>>(sum2, sq2, bn2_g, bn2_b, sb2, 96, 1.f/65536.f);
  k_main<<<dim3(16, 16), 256, 0, stream>>>(xs, conv1_w, conv1_b, sb1, conv2_w, conv2_b, sb2,
                                           conv3_w, conv3_b, wg, hbuf, sum3, sq3);
  k_bnparams<<<2, 256, 0, stream>>>(sum3, sq3, bn3_g, bn3_b, sb3, 32, 1.f/256.f);
  k_scatter<<<16, 256, 0, stream>>>(hbuf, sb3, sidx, xg);
  k_out<<<8192, 256, 0, stream>>>(x_concat, xg, (float*)d_out);
}
